// MultiHeadedAttn_42588895707900
// MI455X (gfx1250) — matrix operand order repaired and checked by reading, not compiled, not run on hardware
//
#include <hip/hip_runtime.h>
#include <hip/hip_bf16.h>

// Problem constants (from reference): B=4, T=2048, D=512, H=8, DK=64
#define BB  4
#define TT  2048
#define DD  512
#define HH  8
#define DKK 64

typedef __attribute__((ext_vector_type(16))) _Float16 v16h;
typedef __attribute__((ext_vector_type(8)))  _Float16 v8h;
typedef __attribute__((ext_vector_type(8)))  float    v8f;
typedef __attribute__((ext_vector_type(4)))  float    f32x4;

#define WMMA_F16(a, b, c) \
  __builtin_amdgcn_wmma_f32_16x16x32_f16(false, (a), false, (b), (short)0, (c), false, false)

// Fast 1-ulp reciprocal (v_rcp_f32) -- avoids the IEEE v_div_scale/v_div_fmas
// sequence the compiler otherwise emits for 1.0f/x.
static __device__ inline float fast_rcp(float x) { return __builtin_amdgcn_rcpf(x); }

// ---------------------------------------------------------------------------
// Fragment loaders following the CDNA5 wave32 WMMA VGPR layouts (ISA 7.12.2).
// A (16x32 f16): lane L holds row L%16; elements e=0..7 -> K = 8*(L/16)+e,
//                e=8..15 -> K = 16+8*(L/16)+(e-8). Two contiguous 8-half runs.
// B (32x16 f16): lane L holds col L%16; elements e=0..7 -> K = 8*(L/16)+e,
//                e=8..15 -> K = 16+8*(L/16)+(e-8). Same as A.
// C/D (16x16 f32): lane L, vgpr v holds (row = v + 8*(L/16), col = L%16).
// ---------------------------------------------------------------------------

static __device__ inline v16h load_a_f32(const float* __restrict__ rowp, int k0, int hs) {
  const f32x4* p0 = (const f32x4*)(rowp + k0 + 8 * hs);
  const f32x4* p1 = (const f32x4*)(rowp + k0 + 16 + 8 * hs);
  f32x4 x0 = p0[0], x1 = p0[1], y0 = p1[0], y1 = p1[1];
  v16h a;
#pragma unroll
  for (int i = 0; i < 4; ++i) {
    a[i]      = (_Float16)x0[i];
    a[4 + i]  = (_Float16)x1[i];
    a[8 + i]  = (_Float16)y0[i];
    a[12 + i] = (_Float16)y1[i];
  }
  return a;
}

static __device__ inline v16h load_b_f32(const float* __restrict__ rowp, int k0, int hs) {
  const f32x4* p0 = (const f32x4*)(rowp + k0 + 8 * hs);
  const f32x4* p1 = (const f32x4*)(rowp + k0 + 16 + 8 * hs);
  v16h b;
#pragma unroll
  for (int j = 0; j < 4; ++j) {
    f32x4 x = (j < 2) ? p0[j] : p1[j - 2];
#pragma unroll
    for (int i = 0; i < 4; ++i) b[4 * j + i] = (_Float16)x[i];
  }
  return b;
}

static __device__ inline v16h load_a_f16(const _Float16* rowp, int k0, int hs) {
  v8h x0 = *(const v8h*)(rowp + k0 + 8 * hs);
  v8h x1 = *(const v8h*)(rowp + k0 + 16 + 8 * hs);
  v16h a;
#pragma unroll
  for (int i = 0; i < 8; ++i) { a[i] = x0[i]; a[8 + i] = x1[i]; }
  return a;
}

static __device__ inline v16h load_b_f16(const _Float16* __restrict__ rowp, int k0, int hs) {
  v8h x0 = *(const v8h*)(rowp + k0 + 8 * hs);
  v8h x1 = *(const v8h*)(rowp + k0 + 16 + 8 * hs);
  v16h b;
#pragma unroll
  for (int i = 0; i < 8; ++i) { b[i] = x0[i]; b[8 + i] = x1[i]; }
  return b;
}

// ---------------------------------------------------------------------------
// Kernel 1: Y = X @ W^T + bias, f32 in -> f16 out, head-split layout.
// VMODE 0: Y[b][h][t][dk]   (row-major over DK)  -- for Q and K
// VMODE 1: Y[b][h][dk][t]   (transposed)         -- for V (so PV B-frags are
//                                                   contiguous along t)
// One wave owns one 16x16 output tile; K-loop of 512/32 = 16 WMMAs.
// ---------------------------------------------------------------------------
template <int VMODE>
__global__ __launch_bounds__(256) void mha_proj_f32_kernel(
    const float* __restrict__ X, const float* __restrict__ W,
    const float* __restrict__ bias, _Float16* __restrict__ Y) {
  const int wid  = (blockIdx.x * blockDim.x + threadIdx.x) >> 5;
  const int lane = threadIdx.x & 31;
  const int r = lane & 15, hs = lane >> 4;
  const int NT = DD / 16;                 // 32 n-tiles
  const int m0 = (wid / NT) * 16;
  const int n0 = (wid % NT) * 16;

  const float* Arow = X + (size_t)(m0 + r) * DD;
  const float* Brow = W + (size_t)(n0 + r) * DD;   // W is [N][K], row-major

  const float bv = bias[n0 + r];
  v8f c;
#pragma unroll
  for (int i = 0; i < 8; ++i) c[i] = bv;

  for (int k0 = 0; k0 < DD; k0 += 32) {
    v16h a = load_a_f32(Arow, k0, hs);
    v16h b = load_b_f32(Brow, k0, hs);
    c = WMMA_F16(a, b, c);
  }

  // Head-split store. 16 | n0 and 16 | DKK, so h/dk decompose is uniform.
  const int n  = n0 + r;
  const int h  = n / DKK, dk = n % DKK;
#pragma unroll
  for (int v = 0; v < 8; ++v) {
    const int m  = m0 + v + 8 * hs;
    const int bb = m / TT, t = m % TT;
    const size_t idx = (VMODE == 0)
        ? ((size_t)((bb * HH + h) * TT + t)) * DKK + dk
        : ((size_t)((bb * HH + h) * DKK + dk)) * TT + t;
    Y[idx] = (_Float16)c[v];
  }
}

// ---------------------------------------------------------------------------
// Kernel 2: fused attention (flash-style online softmax).
// One wave owns (b, h, 16-row q-tile); streams keys in 32-wide chunks.
// Per chunk: 4 score WMMAs, mask+scale, sigmoid stored straight to d_out
// (non-temporal: 512 MB write-once stream must not evict K/V from L2),
// online-softmax update (shfl_xor row reductions), P tile bounced through
// LDS (D-layout scatter -> A-layout b128 gather), 4 PV WMMAs.
// ---------------------------------------------------------------------------
__global__ __launch_bounds__(256) void mha_attn_kernel(
    const _Float16* __restrict__ qbuf, const _Float16* __restrict__ kbuf,
    const _Float16* __restrict__ vbuf, const int* __restrict__ mask,
    float* __restrict__ sig, _Float16* __restrict__ xbuf) {
  __shared__ _Float16 pls[8][16 * 32];    // 1 KB per wave
  const int wslot = threadIdx.x >> 5;
  const int wid   = (blockIdx.x * blockDim.x + threadIdx.x) >> 5;
  const int lane  = threadIdx.x & 31;
  const int r = lane & 15, hs = lane >> 4;

  const int QT = TT / 16;                 // 128 q-tiles per (b,h)
  const int qt = wid % QT;
  const int h  = (wid / QT) % HH;
  const int bb = wid / (QT * HH);
  const int q0 = qt * 16;
  const float scale = 0.125f;             // 1/sqrt(DK)

  const _Float16* qb = qbuf + (size_t)(bb * HH + h) * TT * DKK;
  const _Float16* kb = kbuf + (size_t)(bb * HH + h) * TT * DKK;
  const _Float16* vb = vbuf + (size_t)(bb * HH + h) * DKK * TT;
  float* sigb = sig + (size_t)(bb * HH + h) * TT * TT;

  // Q fragments for both DK halves, reused across all key chunks.
  const v16h aq0 = load_a_f16(qb + (size_t)(q0 + r) * DKK, 0, hs);
  const v16h aq1 = load_a_f16(qb + (size_t)(q0 + r) * DKK, 32, hs);

  v8f o0{}, o1{}, o2{}, o3{};
  float mstat[8], lstat[8];
#pragma unroll
  for (int v = 0; v < 8; ++v) { mstat[v] = -3.0e38f; lstat[v] = 0.0f; }

  for (int tk0 = 0; tk0 < TT; tk0 += 32) {
    // ---- S = Q K^T for two 16-key tiles -------------------------------
    v8f s0{}, s1{};
    {
      v16h bk = load_b_f16(kb + (size_t)(tk0 + r) * DKK, 0, hs);
      s0 = WMMA_F16(aq0, bk, s0);
      bk = load_b_f16(kb + (size_t)(tk0 + r) * DKK, 32, hs);
      s0 = WMMA_F16(aq1, bk, s0);
      bk = load_b_f16(kb + (size_t)(tk0 + 16 + r) * DKK, 0, hs);
      s1 = WMMA_F16(aq0, bk, s1);
      bk = load_b_f16(kb + (size_t)(tk0 + 16 + r) * DKK, 32, hs);
      s1 = WMMA_F16(aq1, bk, s1);
    }

    const int mk0 = mask[bb * TT + tk0 + r];
    const int mk1 = mask[bb * TT + tk0 + 16 + r];

    // ---- scale, mask, sigmoid output (non-temporal stream) ------------
#pragma unroll
    for (int v = 0; v < 8; ++v) {
      float a0 = s0[v] * scale; if (mk0 == 0) a0 = -1.0e9f;
      float a1 = s1[v] * scale; if (mk1 == 0) a1 = -1.0e9f;
      s0[v] = a0; s1[v] = a1;
      const size_t row = (size_t)(q0 + v + 8 * hs) * TT;
      __builtin_nontemporal_store(fast_rcp(1.0f + __expf(-a0)),
                                  sigb + row + tk0 + r);
      __builtin_nontemporal_store(fast_rcp(1.0f + __expf(-a1)),
                                  sigb + row + tk0 + 16 + r);
    }

    // ---- online softmax + stash P (f16) into LDS in D-layout ----------
#pragma unroll
    for (int v = 0; v < 8; ++v) {
      float mx = fmaxf(s0[v], s1[v]);
      mx = fmaxf(mx, __shfl_xor(mx, 1, 32));
      mx = fmaxf(mx, __shfl_xor(mx, 2, 32));
      mx = fmaxf(mx, __shfl_xor(mx, 4, 32));
      mx = fmaxf(mx, __shfl_xor(mx, 8, 32));
      const float mnew = fmaxf(mstat[v], mx);
      const float corr = __expf(mstat[v] - mnew);
      const float p0 = __expf(s0[v] - mnew);
      const float p1 = __expf(s1[v] - mnew);
      float rs = p0 + p1;
      rs += __shfl_xor(rs, 1, 32);
      rs += __shfl_xor(rs, 2, 32);
      rs += __shfl_xor(rs, 4, 32);
      rs += __shfl_xor(rs, 8, 32);
      lstat[v] = lstat[v] * corr + rs;
      mstat[v] = mnew;
      o0[v] *= corr; o1[v] *= corr; o2[v] *= corr; o3[v] *= corr;
      const int row = v + 8 * hs;
      pls[wslot][row * 32 + r]      = (_Float16)p0;
      pls[wslot][row * 32 + 16 + r] = (_Float16)p1;
    }

    // DS ops from one wave execute in order; wait for the scatter stores,
    // then gather the A-layout fragment with b128 LDS reads.
    asm volatile("s_wait_dscnt 0" ::: "memory");
    const v16h ap = load_a_f16(&pls[wslot][0] + r * 32, 0, hs);
    asm volatile("s_wait_dscnt 0" ::: "memory");

    // ---- O += P @ V (V stored transposed: contiguous along t) ---------
    {
      v16h bv0 = load_b_f16(vb + (size_t)(0 * 16 + r) * TT, tk0, hs);
      o0 = WMMA_F16(ap, bv0, o0);
      bv0 = load_b_f16(vb + (size_t)(1 * 16 + r) * TT, tk0, hs);
      o1 = WMMA_F16(ap, bv0, o1);
      bv0 = load_b_f16(vb + (size_t)(2 * 16 + r) * TT, tk0, hs);
      o2 = WMMA_F16(ap, bv0, o2);
      bv0 = load_b_f16(vb + (size_t)(3 * 16 + r) * TT, tk0, hs);
      o3 = WMMA_F16(ap, bv0, o3);
    }
  }

  // ---- finalize: O /= l, store merged-head x (f16, row-major D) --------
#pragma unroll
  for (int v = 0; v < 8; ++v) {
    const float inv = fast_rcp(lstat[v]);
    const int t = q0 + v + 8 * hs;
    _Float16* xr = xbuf + ((size_t)(bb * TT + t)) * DD + h * DKK;
    xr[0 * 16 + r] = (_Float16)(o0[v] * inv);
    xr[1 * 16 + r] = (_Float16)(o1[v] * inv);
    xr[2 * 16 + r] = (_Float16)(o2[v] * inv);
    xr[3 * 16 + r] = (_Float16)(o3[v] * inv);
  }
}

// ---------------------------------------------------------------------------
// Kernel 3: out = x @ Wo^T + bo, f16 activations x f32 weights -> f32 out.
// ---------------------------------------------------------------------------
__global__ __launch_bounds__(256) void mha_proj_out_kernel(
    const _Float16* __restrict__ X, const float* __restrict__ W,
    const float* __restrict__ bias, float* __restrict__ out) {
  const int wid  = (blockIdx.x * blockDim.x + threadIdx.x) >> 5;
  const int lane = threadIdx.x & 31;
  const int r = lane & 15, hs = lane >> 4;
  const int NT = DD / 16;
  const int m0 = (wid / NT) * 16;
  const int n0 = (wid % NT) * 16;

  const _Float16* Arow = X + (size_t)(m0 + r) * DD;
  const float*    Brow = W + (size_t)(n0 + r) * DD;

  const float bv = bias[n0 + r];
  v8f c;
#pragma unroll
  for (int i = 0; i < 8; ++i) c[i] = bv;

  for (int k0 = 0; k0 < DD; k0 += 32) {
    v16h a = load_a_f16(Arow, k0, hs);
    v16h b = load_b_f32(Brow, k0, hs);
    c = WMMA_F16(a, b, c);
  }

#pragma unroll
  for (int v = 0; v < 8; ++v)
    out[(size_t)(m0 + v + 8 * hs) * DD + n0 + r] = c[v];
}

// ---------------------------------------------------------------------------
// Launch: 3 QKV projections -> fused attention -> output projection.
// Workspace: q,k,v (f16 head-split; v transposed) + merged x (f16) = 32 MB.
// d_out = [out (B*T*D f32) | sigmoid_p_attn (B*H*T*T f32)].
// ---------------------------------------------------------------------------
extern "C" void kernel_launch(void* const* d_in, const int* in_sizes, int n_in,
                              void* d_out, int out_size, void* d_ws, size_t ws_size,
                              hipStream_t stream) {
  (void)in_sizes; (void)n_in; (void)out_size; (void)ws_size;
  const float* query = (const float*)d_in[0];
  const float* key_  = (const float*)d_in[1];
  const float* value = (const float*)d_in[2];
  const int*   mask  = (const int*)d_in[3];
  const float* Wq = (const float*)d_in[4];
  const float* bq = (const float*)d_in[5];
  const float* Wk = (const float*)d_in[6];
  const float* bk = (const float*)d_in[7];
  const float* Wv = (const float*)d_in[8];
  const float* bv = (const float*)d_in[9];
  const float* Wo = (const float*)d_in[10];
  const float* bo = (const float*)d_in[11];

  float* out = (float*)d_out;
  float* sig = out + (size_t)BB * TT * DD;

  const size_t NBTD = (size_t)BB * TT * DD;       // 4,194,304 elements
  _Float16* qbuf = (_Float16*)d_ws;
  _Float16* kbuf = qbuf + NBTD;
  _Float16* vbuf = kbuf + NBTD;
  _Float16* xbuf = vbuf + NBTD;

  const dim3 blk(256);
  const int waves_gemm = (BB * TT / 16) * (DD / 16);   // 16384 wave-tiles
  const dim3 grd_gemm(waves_gemm * 32 / 256);          // 2048 blocks

  mha_proj_f32_kernel<0><<<grd_gemm, blk, 0, stream>>>(query, Wq, bq, qbuf);
  mha_proj_f32_kernel<0><<<grd_gemm, blk, 0, stream>>>(key_,  Wk, bk, kbuf);
  mha_proj_f32_kernel<1><<<grd_gemm, blk, 0, stream>>>(value, Wv, bv, vbuf);

  const int waves_attn = BB * HH * (TT / 16);          // 4096 wave-tiles
  const dim3 grd_attn(waves_attn * 32 / 256);          // 512 blocks
  mha_attn_kernel<<<grd_attn, blk, 0, stream>>>(qbuf, kbuf, vbuf, mask, sig, xbuf);

  mha_proj_out_kernel<<<grd_gemm, blk, 0, stream>>>(xbuf, Wo, bo, out);
}
